// LocalAttention_56229711839912
// MI455X (gfx1250) — matrix operand order repaired and checked by reading, not compiled, not run on hardware
//
#include <hip/hip_runtime.h>
#include <hip/hip_bf16.h>

// ---------------------------------------------------------------------------
// Problem constants (from reference): b=64, l=512, n=1024, txt=768, dim=1024
// ---------------------------------------------------------------------------
#define B_   64
#define L_   512
#define N_   1024
#define TXT_ 768
#define DIM_ 1024
#define SCALE_ 0.03125f   // 1024^-0.5

typedef __attribute__((ext_vector_type(16))) __bf16 v16bf;
typedef __attribute__((ext_vector_type(8)))  float  v8f;

// ---------------------------------------------------------------------------
// WMMA fragment builders.
// CDNA5 16-bit A-matrix 16x32 layout (ISA 7.12.2):
//   lane L (0..15): row M=L,   kBase=0 ; lane 16..31: row M=L-16, kBase=8
//   VGPR v=0..3 -> K = kBase + 2v, 2v+1 ; v=4..7 -> K = kBase + 16 + 2(v-4), +1
// CDNA5 16-bit B-matrix 32x16 layout (same K mapping as A):
//   lane L (0..15): col N=L, kBase=0 ; lanes 16..31: kBase=8 (K as for A)
// Both operands here are K-contiguous rows in memory ("NT" GEMM), so each
// lane just reads contiguous pairs from its row pointer (merged into
// global_load_b128 by the compiler) and packs with v_cvt_pk_bf16_f32.
// ---------------------------------------------------------------------------
__device__ __forceinline__ v16bf loadA_f32(const float* __restrict__ row, int laneHi) {
  const int kb = laneHi * 8;
  v16bf a{};
#pragma unroll
  for (int v = 0; v < 8; ++v) {
    const int k = kb + ((v < 4) ? (2 * v) : (16 + 2 * (v - 4)));
    a[2 * v]     = (__bf16)row[k];
    a[2 * v + 1] = (__bf16)row[k + 1];
  }
  return a;
}

__device__ __forceinline__ v16bf loadA_bf16(const __bf16* __restrict__ row, int laneHi) {
  const int kb = laneHi * 8;
  v16bf a{};
#pragma unroll
  for (int v = 0; v < 8; ++v) {
    const int k = kb + ((v < 4) ? (2 * v) : (16 + 2 * (v - 4)));
    a[2 * v]     = row[k];
    a[2 * v + 1] = row[k + 1];
  }
  return a;
}

__device__ __forceinline__ v16bf loadB_f32(const float* __restrict__ row, int laneHi) {
  const int kb = laneHi * 8;
  v16bf b{};
#pragma unroll
  for (int v = 0; v < 8; ++v) {
    const int k = kb + ((v < 4) ? (2 * v) : (16 + 2 * (v - 4)));
    b[2 * v]     = (__bf16)row[k];
    b[2 * v + 1] = (__bf16)row[k + 1];
  }
  return b;
}

__device__ __forceinline__ v8f wmma_bf16(v16bf a, v16bf b, v8f c) {
  // (neg_a, A, neg_b, B, c_mod, C, reuse_a, reuse_b)
  return __builtin_amdgcn_wmma_f32_16x16x32_bf16(false, a, false, b, (short)0, c,
                                                 false, false);
}

// Prefetch distance: 2 K-chunks (64 f32 / 64 bf16 elements) ahead.
#define PF_DIST 64

// ---------------------------------------------------------------------------
// Kernel 1: t = texts @ W_txt^T + b_txt   -> bf16 workspace [32768, 1024]
// One wave per 32x32 output tile; blockDim = (32, 4) = 4 waves.
// ---------------------------------------------------------------------------
__global__ void txt_proj_kernel(const float* __restrict__ texts,
                                const float* __restrict__ Wt,
                                const float* __restrict__ bt,
                                __bf16* __restrict__ tout) {
  const int lane   = threadIdx.x;
  const int laneHi = lane >> 4;
  const int laneLo = lane & 15;
  const int tile   = blockIdx.x * blockDim.y + threadIdx.y;
  const int tn = tile & 31;   // N = 1024 -> 32 tiles
  const int tm = tile >> 5;   // M = 32768 -> 1024 tiles

  const float* a0 = texts + (size_t)(tm * 32 + laneLo) * TXT_;
  const float* a1 = a0 + (size_t)16 * TXT_;
  const float* b0 = Wt + (size_t)(tn * 32 + laneLo) * TXT_;
  const float* b1 = b0 + (size_t)16 * TXT_;

  v8f c00{}, c01{}, c10{}, c11{};
  for (int kc = 0; kc < TXT_; kc += 32) {
    // stream-ahead prefetch of the HBM-resident A panel (global_prefetch_b8)
    __builtin_prefetch(a0 + kc + PF_DIST, 0, 3);
    __builtin_prefetch(a1 + kc + PF_DIST, 0, 3);
    v16bf fa0 = loadA_f32(a0 + kc, laneHi);
    v16bf fa1 = loadA_f32(a1 + kc, laneHi);
    v16bf fb0 = loadB_f32(b0 + kc, laneHi);
    v16bf fb1 = loadB_f32(b1 + kc, laneHi);
    c00 = wmma_bf16(fa0, fb0, c00);
    c01 = wmma_bf16(fa0, fb1, c01);
    c10 = wmma_bf16(fa1, fb0, c10);
    c11 = wmma_bf16(fa1, fb1, c11);
  }

  const int col0 = tn * 32 + laneLo;
  const float bias0 = bt[col0], bias1 = bt[col0 + 16];
#pragma unroll
  for (int r = 0; r < 8; ++r) {
    const int row = tm * 32 + laneHi * 8 + r;
    tout[(size_t)row * DIM_ + col0]             = (__bf16)(c00[r] + bias0);
    tout[(size_t)row * DIM_ + col0 + 16]        = (__bf16)(c01[r] + bias1);
    tout[(size_t)(row + 16) * DIM_ + col0]      = (__bf16)(c10[r] + bias0);
    tout[(size_t)(row + 16) * DIM_ + col0 + 16] = (__bf16)(c11[r] + bias1);
  }
}

// ---------------------------------------------------------------------------
// Kernel 2: im = images @ W_img^T + b_img -> f32 into d_out seg0 [65536, 1024]
// ---------------------------------------------------------------------------
__global__ void img_proj_kernel(const float* __restrict__ images,
                                const float* __restrict__ Wi,
                                const float* __restrict__ bi,
                                float* __restrict__ imout) {
  const int lane   = threadIdx.x;
  const int laneHi = lane >> 4;
  const int laneLo = lane & 15;
  const int tile   = blockIdx.x * blockDim.y + threadIdx.y;
  const int tn = tile & 31;   // N = 1024 -> 32 tiles
  const int tm = tile >> 5;   // M = 65536 -> 2048 tiles

  const float* a0 = images + (size_t)(tm * 32 + laneLo) * DIM_;
  const float* a1 = a0 + (size_t)16 * DIM_;
  const float* b0 = Wi + (size_t)(tn * 32 + laneLo) * DIM_;
  const float* b1 = b0 + (size_t)16 * DIM_;

  v8f c00{}, c01{}, c10{}, c11{};
  for (int kc = 0; kc < DIM_; kc += 32) {
    __builtin_prefetch(a0 + kc + PF_DIST, 0, 3);
    __builtin_prefetch(a1 + kc + PF_DIST, 0, 3);
    v16bf fa0 = loadA_f32(a0 + kc, laneHi);
    v16bf fa1 = loadA_f32(a1 + kc, laneHi);
    v16bf fb0 = loadB_f32(b0 + kc, laneHi);
    v16bf fb1 = loadB_f32(b1 + kc, laneHi);
    c00 = wmma_bf16(fa0, fb0, c00);
    c01 = wmma_bf16(fa0, fb1, c01);
    c10 = wmma_bf16(fa1, fb0, c10);
    c11 = wmma_bf16(fa1, fb1, c11);
  }

  const int col0 = tn * 32 + laneLo;
  const float bias0 = bi[col0], bias1 = bi[col0 + 16];
#pragma unroll
  for (int r = 0; r < 8; ++r) {
    const int row = tm * 32 + laneHi * 8 + r;
    imout[(size_t)row * DIM_ + col0]             = c00[r] + bias0;
    imout[(size_t)row * DIM_ + col0 + 16]        = c01[r] + bias1;
    imout[(size_t)(row + 16) * DIM_ + col0]      = c10[r] + bias0;
    imout[(size_t)(row + 16) * DIM_ + col0 + 16] = c11[r] + bias1;
  }
}

// ---------------------------------------------------------------------------
// Kernel 3: dots[b,i,j] = SCALE * sum_d t[b,i,d] * im[b,j,d]
// A = t (bf16, ws), B = im (f32 in d_out seg0, converted in-register).
// grid = (128, 64): 512 tiles/batch, 4 waves/block; blockIdx.y = batch.
// ---------------------------------------------------------------------------
__global__ void dots_kernel(const __bf16* __restrict__ t,
                            const float* __restrict__ im,
                            float* __restrict__ dots) {
  const int lane   = threadIdx.x;
  const int laneHi = lane >> 4;
  const int laneLo = lane & 15;
  const int batch  = blockIdx.y;
  const int tile   = blockIdx.x * blockDim.y + threadIdx.y; // 0..511
  const int tn = tile & 31;   // N = 1024 -> 32 tiles
  const int tm = tile >> 5;   // M = 512  -> 16 tiles

  const __bf16* abase = t  + (size_t)batch * L_ * DIM_;
  const float*  bbase = im + (size_t)batch * N_ * DIM_;
  const __bf16* a0 = abase + (size_t)(tm * 32 + laneLo) * DIM_;
  const __bf16* a1 = a0 + (size_t)16 * DIM_;
  const float*  b0 = bbase + (size_t)(tn * 32 + laneLo) * DIM_;
  const float*  b1 = b0 + (size_t)16 * DIM_;

  v8f c00{}, c01{}, c10{}, c11{};
  for (int kc = 0; kc < DIM_; kc += 32) {
    __builtin_prefetch(b0 + kc + PF_DIST, 0, 3);
    __builtin_prefetch(b1 + kc + PF_DIST, 0, 3);
    v16bf fa0 = loadA_bf16(a0 + kc, laneHi);
    v16bf fa1 = loadA_bf16(a1 + kc, laneHi);
    v16bf fb0 = loadB_f32(b0 + kc, laneHi);
    v16bf fb1 = loadB_f32(b1 + kc, laneHi);
    c00 = wmma_bf16(fa0, fb0, c00);
    c01 = wmma_bf16(fa0, fb1, c01);
    c10 = wmma_bf16(fa1, fb0, c10);
    c11 = wmma_bf16(fa1, fb1, c11);
  }

  float* dst = dots + (size_t)batch * L_ * N_;
  const int col0 = tn * 32 + laneLo;
#pragma unroll
  for (int r = 0; r < 8; ++r) {
    const int row = tm * 32 + laneHi * 8 + r;
    dst[(size_t)row * N_ + col0]             = c00[r] * SCALE_;
    dst[(size_t)row * N_ + col0 + 16]        = c01[r] * SCALE_;
    dst[(size_t)(row + 16) * N_ + col0]      = c10[r] * SCALE_;
    dst[(size_t)(row + 16) * N_ + col0 + 16] = c11[r] * SCALE_;
  }
}

// ---------------------------------------------------------------------------
// Kernel 4: per-batch flat softmax statistics over l*n = 524288 elements.
// One block (1024 threads = 32 waves) per batch. stats[b]=max, stats[64+b]=1/sum.
// ---------------------------------------------------------------------------
__global__ void softmax_stats_kernel(const float* __restrict__ dots,
                                     float* __restrict__ stats) {
  __shared__ float red[1024];
  const int b   = blockIdx.x;
  const int tid = threadIdx.x;
  const float* p = dots + (size_t)b * (L_ * N_);

  float m = -3.4e38f;
  for (int i = tid; i < L_ * N_; i += 1024) m = fmaxf(m, p[i]);
  red[tid] = m;
  __syncthreads();
  for (int s = 512; s > 0; s >>= 1) {
    if (tid < s) red[tid] = fmaxf(red[tid], red[tid + s]);
    __syncthreads();
  }
  const float mx = red[0];
  __syncthreads();

  float sum = 0.f;
  for (int i = tid; i < L_ * N_; i += 1024) sum += expf(p[i] - mx);
  red[tid] = sum;
  __syncthreads();
  for (int s = 512; s > 0; s >>= 1) {
    if (tid < s) red[tid] += red[tid + s];
    __syncthreads();
  }
  if (tid == 0) {
    stats[b]       = mx;
    stats[B_ + b]  = 1.0f / red[0];
  }
}

// ---------------------------------------------------------------------------
// Kernel 5: normalize dots in place; local_attn[b,j] = sum_i norm_dots[b,i,j].
// grid = (4, 64); thread owns one column j, loops over i (coalesced over j).
// ---------------------------------------------------------------------------
__global__ void normalize_kernel(float* __restrict__ dots,
                                 const float* __restrict__ stats,
                                 float* __restrict__ la) {
  const int b = blockIdx.y;
  const int j = blockIdx.x * blockDim.x + threadIdx.x;
  const float mx  = stats[b];
  const float inv = stats[B_ + b];
  float* p = dots + (size_t)b * (L_ * N_) + j;
  float acc = 0.f;
  for (int i = 0; i < L_; ++i) {
    const float v = expf(p[(size_t)i * N_] - mx) * inv;
    p[(size_t)i * N_] = v;
    acc += v;
  }
  la[b * N_ + j] = acc;
}

// ---------------------------------------------------------------------------
// Kernel 6: out[b,j,:] = local_attn[b,j] * im[b,j,:]   (in place on seg0)
// ---------------------------------------------------------------------------
__global__ void scale_kernel(float* __restrict__ out, const float* __restrict__ la) {
  const size_t idx = (size_t)blockIdx.x * blockDim.x + threadIdx.x;
  out[idx] *= la[idx >> 10];   // row index = b*1024 + j
}

// ---------------------------------------------------------------------------
extern "C" void kernel_launch(void* const* d_in, const int* in_sizes, int n_in,
                              void* d_out, int out_size, void* d_ws, size_t ws_size,
                              hipStream_t stream) {
  (void)in_sizes; (void)n_in; (void)out_size; (void)ws_size;
  const float* images = (const float*)d_in[0];  // [64,1024,1024]
  const float* texts  = (const float*)d_in[1];  // [64,512,768]
  const float* W_txt  = (const float*)d_in[2];  // [1024,768]
  const float* b_txt  = (const float*)d_in[3];  // [1024]
  const float* W_img  = (const float*)d_in[4];  // [1024,1024]
  const float* b_img  = (const float*)d_in[5];  // [1024]

  float* out  = (float*)d_out;                       // [64,1024,1024]
  float* norm = out + (size_t)B_ * N_ * DIM_;        // [64,512,1024]
  float* la   = norm + (size_t)B_ * L_ * N_;         // [64,1024]

  __bf16* t_ws = (__bf16*)d_ws;                      // [32768,1024] bf16 (64 MB)
  float* stats = (float*)((char*)d_ws + (size_t)B_ * L_ * DIM_ * sizeof(__bf16));

  const dim3 blk(32, 4);  // 4 waves per block (wave32)
  // 1) t projection: 1024*32 tiles / 4 waves = 8192 blocks
  txt_proj_kernel<<<8192, blk, 0, stream>>>(texts, W_txt, b_txt, t_ws);
  // 2) im projection: 2048*32 tiles / 4 = 16384 blocks (writes d_out seg0)
  img_proj_kernel<<<16384, blk, 0, stream>>>(images, W_img, b_img, out);
  // 3) batched dots: 512 tiles/batch / 4 = 128 blocks x 64 batches
  dots_kernel<<<dim3(128, 64), blk, 0, stream>>>(t_ws, out, norm);
  // 4) per-batch softmax stats
  softmax_stats_kernel<<<B_, 1024, 0, stream>>>(norm, stats);
  // 5) normalize + column-sum -> local_attn
  normalize_kernel<<<dim3(N_ / 256, B_), 256, 0, stream>>>(norm, stats, la);
  // 6) scale im by local_attn in place -> out
  scale_kernel<<<(B_ * N_ * DIM_) / 256, 256, 0, stream>>>(out, la);
}
